// MiniGPTModel_84928683311708
// MI455X (gfx1250) — matrix operand order repaired and checked by reading, not compiled, not run on hardware
//
#include <hip/hip_runtime.h>
#include <hip/hip_bf16.h>

#define BATCH  2
#define WINDOW 2048
#define DIM    1024
#define VOCAB  32000

typedef __bf16 bf16;
typedef __attribute__((ext_vector_type(16))) __bf16 v16bf;
typedef __attribute__((ext_vector_type(8)))  __bf16 v8bf;
typedef __attribute__((ext_vector_type(8)))  float  v8f;

// ---------------------------------------------------------------------------
// WMMA helpers (CDNA5 16x16x32 bf16, f32 accumulate)
// ---------------------------------------------------------------------------
__device__ __forceinline__ v8f wmma_bf16(v16bf a, v16bf b, v8f c) {
    // 8 args: (neg_a, A, neg_b, B, c_mod, C, reuse_a, reuse_b)
    return __builtin_amdgcn_wmma_f32_16x16x32_bf16(
        false, a, false, b, (short)0, c, false, false);
}

// A-matrix fragment, 16x32 bf16 (M x K), A stored row-major [M][lda].
// Contiguous k order used here (not AMD's table order): lane L holds row M = L&15; lane half selects K half:
//   lanes 0-15 : K = 0..15
//   lanes 16-31: K = 16..31
__device__ __forceinline__ v16bf load_a_frag(const bf16* A, int lda,
                                             int m0, int k0, int lane) {
    int row = m0 + (lane & 15);
    int kb  = (lane >> 4) * 16;
    const bf16* p = A + (long)row * lda + k0 + kb;
    v8bf lo = *(const v8bf*)(p);        // K = kb .. kb+7
    v8bf hi = *(const v8bf*)(p + 8);    // K = kb+8 .. kb+15
    v16bf r;
#pragma unroll
    for (int i = 0; i < 8; ++i) { r[i] = lo[i]; r[i + 8] = hi[i]; }
    return r;
}

// B-matrix fragment, 32x16 bf16 (K x N), stored TRANSPOSED row-major as
// Bt[n][k] with leading dim ldb. Lane L holds column N = L&15 with
// K = (L>>4)*16 .. +15 contiguous -> single 32-byte load.
__device__ __forceinline__ v16bf load_b_frag(const bf16* Bt, int ldb,
                                             int n0, int k0, int lane) {
    int col = n0 + (lane & 15);
    int kb  = (lane >> 4) * 16;
    return *(const v16bf*)(Bt + (long)col * ldb + k0 + kb);
}

// C/D f32 fragment: element i of v8f is (M = i + 8*(lane>>4), N = lane&15).

// ---------------------------------------------------------------------------
// gfx1250 async copy global -> LDS (16B per lane), tracked by ASYNCcnt
// ---------------------------------------------------------------------------
__device__ __forceinline__ void async_copy_b128(const void* gsrc, void* lds_dst) {
    unsigned lds_addr = (unsigned)(unsigned long long)lds_dst;  // low 32b = LDS byte addr
    unsigned long long gaddr = (unsigned long long)gsrc;
    asm volatile("global_load_async_to_lds_b128 %0, %1, off"
                 :: "v"(lds_addr), "v"(gaddr) : "memory");
}

__device__ __forceinline__ void wait_asynccnt0() {
#if __has_builtin(__builtin_amdgcn_s_wait_asynccnt)
    __builtin_amdgcn_s_wait_asynccnt(0);
#else
    asm volatile("s_wait_asynccnt 0x0" ::: "memory");
#endif
}

// ---------------------------------------------------------------------------
// 1) Embedding: h_bf16[row][d] = bf16(tok_emb[x[row]][d] + pos_emb[s][d])
// ---------------------------------------------------------------------------
__global__ void k_embed(const int* __restrict__ x,
                        const float* __restrict__ tok,
                        const float* __restrict__ pos,
                        bf16* __restrict__ h) {
    int row = blockIdx.x;                 // 0 .. B*S-1
    int s   = row & (WINDOW - 1);
    int t   = x[row];
    const float* te = tok + (long)t * DIM;
    const float* pe = pos + (long)s * DIM;
    bf16* o = h + (long)row * DIM;
    for (int d = threadIdx.x; d < DIM; d += blockDim.x)
        o[d] = (bf16)(te[d] + pe[d]);
}

// ---------------------------------------------------------------------------
// 2) Weight prep: W[k][n] (f32, KxN row-major) -> Wt[n][k] (bf16)
// ---------------------------------------------------------------------------
__global__ void k_transpose_bf16(const float* __restrict__ W,
                                 bf16* __restrict__ Wt, int K, int N) {
    long idx = (long)blockIdx.x * blockDim.x + threadIdx.x;
    long total = (long)K * N;
    if (idx >= total) return;
    int n = (int)(idx % N);
    int k = (int)(idx / N);
    Wt[(long)n * K + k] = (bf16)W[idx];
}

// ---------------------------------------------------------------------------
// 3) QKV GEMM: [4096 x 1024] = h @ w + b. One wave per 16x64 tile.
//    z = 0/1/2 -> q / k / v. v written transposed: vT[b][d][s].
// ---------------------------------------------------------------------------
__global__ void k_gemm_qkv(const bf16* __restrict__ h,
                           const bf16* __restrict__ wqT,
                           const bf16* __restrict__ wkT,
                           const bf16* __restrict__ wvT,
                           const float* __restrict__ bq,
                           const float* __restrict__ bk,
                           const float* __restrict__ bv,
                           bf16* __restrict__ q,
                           bf16* __restrict__ k,
                           bf16* __restrict__ vT) {
    int which = blockIdx.z;
    const bf16*  Bt   = (which == 0) ? wqT : (which == 1) ? wkT : wvT;
    const float* bias = (which == 0) ? bq  : (which == 1) ? bk  : bv;
    int m0 = blockIdx.x * 16;
    int n0 = blockIdx.y * 64;
    int lane = threadIdx.x;

    v8f acc[4] = {};
    for (int k0 = 0; k0 < DIM; k0 += 32) {
        v16bf a = load_a_frag(h, DIM, m0, k0, lane);
#pragma unroll
        for (int j = 0; j < 4; ++j) {
            v16bf b = load_b_frag(Bt, DIM, n0 + j * 16, k0, lane);
            acc[j] = wmma_bf16(a, b, acc[j]);
        }
    }

    int mrow = (lane >> 4) * 8;
    int nc   = lane & 15;
#pragma unroll
    for (int j = 0; j < 4; ++j) {
        int n = n0 + j * 16 + nc;
        float bb = bias[n];
#pragma unroll
        for (int i = 0; i < 8; ++i) {
            int m = m0 + mrow + i;
            bf16 val = (bf16)(acc[j][i] + bb);
            if (which == 0)      q[(long)m * DIM + n] = val;
            else if (which == 1) k[(long)m * DIM + n] = val;
            else {
                int bat = m >> 11, s = m & (WINDOW - 1);
                vT[((long)bat * DIM + n) * WINDOW + s] = val;
            }
        }
    }
}

// ---------------------------------------------------------------------------
// 4) Scores: sc[b][s][t] = q[s]·k[t], causal-masked with -inf.
//    Fully-masked 16x16 tiles skip their WMMA work (uniform branch).
// ---------------------------------------------------------------------------
__global__ void k_scores(const bf16* __restrict__ q,
                         const bf16* __restrict__ kmat,
                         float* __restrict__ scores) {
    int b  = blockIdx.z;
    int s0 = blockIdx.x * 16;
    int t0 = blockIdx.y * 64;
    int lane = threadIdx.x;
    const bf16* qb = q    + (long)b * WINDOW * DIM;
    const bf16* kb = kmat + (long)b * WINDOW * DIM;
    float* sc = scores + (long)b * WINDOW * WINDOW;

    v8f acc[4] = {};
    if (t0 <= s0 + 15) {                       // at least one live tile
        for (int d0 = 0; d0 < DIM; d0 += 32) {
            v16bf a = load_a_frag(qb, DIM, s0, d0, lane);
#pragma unroll
            for (int j = 0; j < 4; ++j) {
                if (t0 + j * 16 <= s0 + 15) {  // uniform across wave
                    v16bf bfr = load_b_frag(kb, DIM, t0 + j * 16, d0, lane);
                    acc[j] = wmma_bf16(a, bfr, acc[j]);
                }
            }
        }
    }

    int mrow = (lane >> 4) * 8;
    int nc   = lane & 15;
#pragma unroll
    for (int j = 0; j < 4; ++j) {
        int t = t0 + j * 16 + nc;
#pragma unroll
        for (int i = 0; i < 8; ++i) {
            int s = s0 + mrow + i;
            sc[(long)s * WINDOW + t] = (t <= s) ? acc[j][i] : -__builtin_inff();
        }
    }
}

// ---------------------------------------------------------------------------
// 5) Row softmax: f32 scores -> bf16 probs. One 256-thread block per row.
// ---------------------------------------------------------------------------
__global__ void k_softmax(const float* __restrict__ sc,
                          bf16* __restrict__ p) {
    __shared__ float sm[256];
    long row = blockIdx.x;                      // b*S + s
    const float* r = sc + row * WINDOW;
    bf16* pr = p + row * WINDOW;
    int tid = threadIdx.x;

    float m = -__builtin_inff();
    for (int t = tid; t < WINDOW; t += 256) m = fmaxf(m, r[t]);
    sm[tid] = m; __syncthreads();
    for (int off = 128; off > 0; off >>= 1) {
        if (tid < off) sm[tid] = fmaxf(sm[tid], sm[tid + off]);
        __syncthreads();
    }
    float rowmax = sm[0]; __syncthreads();

    float sum = 0.f;
    for (int t = tid; t < WINDOW; t += 256) sum += __expf(r[t] - rowmax);
    sm[tid] = sum; __syncthreads();
    for (int off = 128; off > 0; off >>= 1) {
        if (tid < off) sm[tid] += sm[tid + off];
        __syncthreads();
    }
    float inv = 1.f / sm[0];

    for (int t = tid; t < WINDOW; t += 256)
        pr[t] = (bf16)(__expf(r[t] - rowmax) * inv);
}

// ---------------------------------------------------------------------------
// 6) attn @ V: o[s][d] = sum_t p[s][t] * v[t][d]. K-loop truncated causally
//    (probs past s are exactly 0, so rounding kmax up to 32 is safe).
// ---------------------------------------------------------------------------
__global__ void k_attn(const bf16* __restrict__ p,
                       const bf16* __restrict__ vT,
                       bf16* __restrict__ o) {
    int b  = blockIdx.z;
    int s0 = blockIdx.x * 16;
    int n0 = blockIdx.y * 64;
    int lane = threadIdx.x;
    const bf16* pb = p  + (long)b * WINDOW * WINDOW;
    const bf16* vb = vT + (long)b * DIM * WINDOW;

    v8f acc[4] = {};
    int kmax = (s0 + 16 + 31) & ~31;
    for (int k0 = 0; k0 < kmax; k0 += 32) {
        v16bf a = load_a_frag(pb, WINDOW, s0, k0, lane);
#pragma unroll
        for (int j = 0; j < 4; ++j) {
            v16bf bfr = load_b_frag(vb, WINDOW, n0 + j * 16, k0, lane);
            acc[j] = wmma_bf16(a, bfr, acc[j]);
        }
    }

    int mrow = (lane >> 4) * 8;
    int nc   = lane & 15;
#pragma unroll
    for (int j = 0; j < 4; ++j) {
        int n = n0 + j * 16 + nc;
#pragma unroll
        for (int i = 0; i < 8; ++i) {
            int m = (b << 11) + s0 + mrow + i;    // global row b*S+s
            o[(long)m * DIM + n] = (bf16)acc[j][i];
        }
    }
}

// ---------------------------------------------------------------------------
// 7) Logits: [4096 x 32000] = o @ wo + bo  (dominant GEMM, 268 GFLOP).
//    Workgroup-tiled: 8 waves, block tile 128x128, K-step 64, operands
//    double-buffered in LDS via async global->LDS copies (ASYNCcnt).
//    Cuts L2 operand traffic ~8x vs the 1-wave/16x64 version.
// ---------------------------------------------------------------------------
#define BM 128
#define BN 128
#define BK 64

__global__ __launch_bounds__(256) void k_logits_tiled(
        const bf16* __restrict__ o,
        const bf16* __restrict__ woT,
        const float* __restrict__ bo,
        float* __restrict__ out) {
    __shared__ bf16 Asm[2][BM * BK];      // [row][k]  16 KB each
    __shared__ bf16 Bsm[2][BN * BK];      // [col][k]  16 KB each

    int tid  = threadIdx.x;
    int wave = tid >> 5;
    int lane = tid & 31;
    int m0 = blockIdx.x * BM;
    int n0 = blockIdx.y * BN;

    // stage one BK-deep slice of A and B into LDS buffer `buf`
    // (each thread moves 4x16B of A and 4x16B of B)
    auto stage = [&](int buf, int k0) {
#pragma unroll
        for (int c = 0; c < 4; ++c) {
            int oe = (wave * 4 + c) * 256 + lane * 8;   // flat bf16 offset in tile
            int r  = oe >> 6;                           // tile row (64 elems/row)
            int kk = oe & 63;
            async_copy_b128(o   + (long)(m0 + r) * DIM + k0 + kk, &Asm[buf][oe]);
            async_copy_b128(woT + (long)(n0 + r) * DIM + k0 + kk, &Bsm[buf][oe]);
        }
    };

    v8f acc[8] = {};
    stage(0, 0);

    for (int k0 = 0; k0 < DIM; k0 += BK) {
        int buf = (k0 / BK) & 1;
        wait_asynccnt0();      // this wave's copies into `buf` are complete
        __syncthreads();       // all waves' copies complete; prev compute done
        if (k0 + BK < DIM) stage(buf ^ 1, k0 + BK);   // overlap with compute

#pragma unroll
        for (int kc = 0; kc < BK; kc += 32) {
            v16bf a = load_a_frag(&Asm[buf][0], BK, wave * 16, kc, lane);
#pragma unroll
            for (int j = 0; j < 8; ++j) {
                v16bf b = load_b_frag(&Bsm[buf][0], BK, j * 16, kc, lane);
                acc[j] = wmma_bf16(a, b, acc[j]);
            }
        }
    }

    int mrow = (lane >> 4) * 8;
    int nc   = lane & 15;
#pragma unroll
    for (int j = 0; j < 8; ++j) {
        int n = n0 + j * 16 + nc;
        float bb = bo[n];
#pragma unroll
        for (int i = 0; i < 8; ++i) {
            int m = m0 + wave * 16 + mrow + i;
            out[(long)m * VOCAB + n] = acc[j][i] + bb;
        }
    }
}

// ---------------------------------------------------------------------------
// Launch
// ---------------------------------------------------------------------------
extern "C" void kernel_launch(void* const* d_in, const int* in_sizes, int n_in,
                              void* d_out, int out_size, void* d_ws, size_t ws_size,
                              hipStream_t stream) {
    const int*   x    = (const int*)  d_in[0];
    const float* tok  = (const float*)d_in[1];
    const float* pos  = (const float*)d_in[2];
    const float* wq   = (const float*)d_in[3];
    const float* bq   = (const float*)d_in[4];
    const float* wk   = (const float*)d_in[5];
    const float* bk   = (const float*)d_in[6];
    const float* wv   = (const float*)d_in[7];
    const float* bv   = (const float*)d_in[8];
    const float* wo   = (const float*)d_in[9];
    const float* bo   = (const float*)d_in[10];
    float* out = (float*)d_out;

    // workspace carve-up (256B aligned)
    char* ws = (char*)d_ws;
    size_t off = 0;
    auto carve = [&](size_t bytes) -> char* {
        char* p = ws + off;
        off = (off + bytes + 255) & ~(size_t)255;
        return p;
    };
    const long MR = (long)BATCH * WINDOW;     // 4096 rows
    bf16* h_bf   = (bf16*)carve(MR * DIM * 2);
    bf16* wqT    = (bf16*)carve((long)DIM * DIM * 2);
    bf16* wkT    = (bf16*)carve((long)DIM * DIM * 2);
    bf16* wvT    = (bf16*)carve((long)DIM * DIM * 2);
    bf16* woT    = (bf16*)carve((long)VOCAB * DIM * 2);
    bf16* qb     = (bf16*)carve(MR * DIM * 2);
    bf16* kb     = (bf16*)carve(MR * DIM * 2);
    bf16* vT     = (bf16*)carve(MR * DIM * 2);          // [b][d][s]
    float* sc    = (float*)carve((long)BATCH * WINDOW * WINDOW * 4);
    bf16* probs  = (bf16*)carve((long)BATCH * WINDOW * WINDOW * 2);
    bf16* attn_o = (bf16*)carve(MR * DIM * 2);
    (void)ws_size; (void)in_sizes; (void)n_in; (void)out_size;

    // 1) embeddings -> bf16 activations
    k_embed<<<(int)MR, 256, 0, stream>>>(x, tok, pos, h_bf);

    // 2) weight transpose+convert (B-fragment friendly [N][K] bf16)
    {
        long t = (long)DIM * DIM;
        int blocks = (int)((t + 255) / 256);
        k_transpose_bf16<<<blocks, 256, 0, stream>>>(wq, wqT, DIM, DIM);
        k_transpose_bf16<<<blocks, 256, 0, stream>>>(wk, wkT, DIM, DIM);
        k_transpose_bf16<<<blocks, 256, 0, stream>>>(wv, wvT, DIM, DIM);
        long to = (long)DIM * VOCAB;
        k_transpose_bf16<<<(int)((to + 255) / 256), 256, 0, stream>>>(wo, woT, DIM, VOCAB);
    }

    // 3) q,k,v projections (one wave per 16x64 tile)
    k_gemm_qkv<<<dim3(MR / 16, DIM / 64, 3), 32, 0, stream>>>(
        h_bf, wqT, wkT, wvT, bq, bk, bv, qb, kb, vT);

    // 4) causal scores
    k_scores<<<dim3(WINDOW / 16, WINDOW / 64, BATCH), 32, 0, stream>>>(qb, kb, sc);

    // 5) softmax -> bf16 probs
    k_softmax<<<(int)MR, 256, 0, stream>>>(sc, probs);

    // 6) attention output
    k_attn<<<dim3(WINDOW / 16, DIM / 64, BATCH), 32, 0, stream>>>(probs, vT, attn_o);

    // 7) logits: LDS-tiled, async-staged, double-buffered (dominant GEMM)
    k_logits_tiled<<<dim3(BATCH * WINDOW / BM, VOCAB / BN), 256, 0, stream>>>(
        attn_o, woT, bo, out);
}
